// WindowAttentionBlock_4741643894858
// MI455X (gfx1250) — compile-verified, matrix operand order checked by reading, not run on hardware
//
#include <hip/hip_runtime.h>
#include <hip/hip_bf16.h>
#include <math.h>

// ---------------- problem constants ----------------
#define B_    32
#define HH    56
#define WWID  56
#define WS_   7
#define SS_   3
#define NH_   6
#define C_    192
#define T_    3136           // HH*WWID
#define HD_   32             // C/NH
#define NWI   64             // windows per image
#define TW_   49             // tokens per window
#define NWIN  2048           // B_*NWI
#define MTOT  100352         // NWIN*TW_
#define SCALE_ 0.17677669529663687f  // 32^-0.5

typedef __attribute__((ext_vector_type(16))) __bf16 v16bf;
typedef __attribute__((ext_vector_type(8)))  float  v8f;
typedef int v4i __attribute__((vector_size(16)));
typedef __attribute__((address_space(1))) v4i as1_v4i;
typedef __attribute__((address_space(3))) v4i as3_v4i;

union Frag { uint4 q[2]; v16bf v; };

#if defined(__has_builtin)
#if __has_builtin(__builtin_amdgcn_global_load_async_to_lds_b128) && \
    __has_builtin(__builtin_amdgcn_s_wait_asynccnt)
#define USE_ASYNC 1
#endif
#endif
#ifndef USE_ASYNC
#define USE_ASYNC 0
#endif

static __device__ __forceinline__ unsigned short f32_bf16(float f) {
  union { float f; unsigned u; } v; v.f = f;
  unsigned r = (v.u + 0x7FFFu + ((v.u >> 16) & 1u)) >> 16;
  return (unsigned short)r;
}

// 16-byte global->LDS copy: async (ASYNCcnt, no VGPR staging) when available.
static __device__ __forceinline__ void cp16(const unsigned short* g, unsigned short* l) {
#if USE_ASYNC
  __builtin_amdgcn_global_load_async_to_lds_b128(
      (as1_v4i*)g, (as3_v4i*)l, 0, 0);
#else
  *(uint4*)l = *(const uint4*)g;
#endif
}

// ---------------- weight transpose + bf16 convert: wt[n*K+k] = w[k*N+n] -----
__global__ void wconv_kernel(const float* __restrict__ w,
                             unsigned short* __restrict__ wt, int K, int N) {
  int o = blockIdx.x * 256 + threadIdx.x;
  if (o >= K * N) return;
  int n = o / K, k = o - n * K;
  wt[o] = f32_bf16(w[(size_t)k * N + n]);
}

// ---------------- LayerNorm (one wave per row), optional shift+window gather -
__global__ void __launch_bounds__(256)
ln_kernel(const float* __restrict__ x, const float* __restrict__ g,
          const float* __restrict__ b, unsigned short* __restrict__ out,
          int permute) {
  int tid  = threadIdx.x;
  int lane = tid & 31, wid = tid >> 5;
  int rd = blockIdx.x * 8 + wid;          // destination row (window domain if permute)
  int src;
  if (permute) {
    int bimg = rd / (NWI * TW_);
    int rem  = rd - bimg * (NWI * TW_);
    int win = rem / TW_, t = rem - win * TW_;
    int wh = win >> 3, ww = win & 7;
    int ih = t / WS_, iw = t - ih * WS_;
    int oh = (wh * WS_ + ih + SS_) % HH;   // cyclic shift -SS => gather +SS
    int ow = (ww * WS_ + iw + SS_) % WWID;
    src = bimg * T_ + oh * WWID + ow;
  } else {
    src = rd;
  }
  const float* row = x + (size_t)src * C_;
  float vals[6];
  float s = 0.f;
#pragma unroll
  for (int e = 0; e < 6; e++) { vals[e] = row[e * 32 + lane]; s += vals[e]; }
#pragma unroll
  for (int off = 16; off > 0; off >>= 1) s += __shfl_xor(s, off, 32);
  float mean = s * (1.f / C_);
  float vs = 0.f;
#pragma unroll
  for (int e = 0; e < 6; e++) { float d = vals[e] - mean; vs += d * d; }
#pragma unroll
  for (int off = 16; off > 0; off >>= 1) vs += __shfl_xor(vs, off, 32);
  float rstd = rsqrtf(vs * (1.f / C_) + 1e-5f);
  unsigned short* orow = out + (size_t)rd * C_;
#pragma unroll
  for (int e = 0; e < 6; e++) {
    int c = e * 32 + lane;
    orow[c] = f32_bf16((vals[e] - mean) * rstd * g[c] + b[c]);
  }
}

// ---------------- async tile stage: A 128x32 (2 chunks/thr) + W 64x32 -------
static __device__ __forceinline__ void
issue_tile(const unsigned short* __restrict__ A, const unsigned short* __restrict__ WT,
           int K, int rowBase, int nBase, int k0,
           unsigned short* dA, unsigned short* dW, int tid) {
  {
    int r0 = tid >> 2, cc0 = tid & 3;
    cp16(A + (size_t)(rowBase + r0) * K + k0 + cc0 * 8, dA + r0 * 32 + cc0 * 8);
    int c1 = tid + 256;
    int r1 = c1 >> 2, cc1 = c1 & 3;
    cp16(A + (size_t)(rowBase + r1) * K + k0 + cc1 * 8, dA + r1 * 32 + cc1 * 8);
  }
  {
    int r = tid >> 2, cc = tid & 3;
    cp16(WT + (size_t)(nBase + r) * K + k0 + cc * 8, dW + r * 32 + cc * 8);
  }
}

// ---------------- WMMA GEMM mainloop: 128x64 tile, K-step 32, dbl-buffered --
// A: [M,K] bf16 row-major. WT: [N,K] bf16 row-major (i.e. W transposed).
static __device__ __forceinline__ void
gemm_main(const unsigned short* __restrict__ A, const unsigned short* __restrict__ WT,
          int K, int rowBase, int nBase,
          unsigned short* ldsA, unsigned short* ldsW, v8f acc[4]) {
  const int tid = threadIdx.x;
  const int lane = tid & 31, wid = tid >> 5;
  const int fr = lane & 15, fh = lane >> 4;
#pragma unroll
  for (int j = 0; j < 4; j++)
    for (int e = 0; e < 8; e++) acc[j][e] = 0.f;

  issue_tile(A, WT, K, rowBase, nBase, 0, ldsA, ldsW, tid);

  int buf = 0;
  for (int k0 = 0; k0 < K; k0 += 32, buf ^= 1) {
    if (k0 + 32 < K) {
      issue_tile(A, WT, K, rowBase, nBase, k0 + 32,
                 ldsA + (buf ^ 1) * (128 * 32), ldsW + (buf ^ 1) * (64 * 32), tid);
#if USE_ASYNC
      __builtin_amdgcn_s_wait_asynccnt(3);   // tile k0 fully landed (ours)
#endif
    } else {
#if USE_ASYNC
      __builtin_amdgcn_s_wait_asynccnt(0);
#endif
    }
    __syncthreads();                          // publish tile k0 across waves
    const unsigned short* cA = ldsA + buf * (128 * 32);
    const unsigned short* cW = ldsW + buf * (64 * 32);
    Frag a, bfr[4];
    a.q[0] = *(const uint4*)(cA + (wid * 16 + fr) * 32 + 8 * fh);
    a.q[1] = *(const uint4*)(cA + (wid * 16 + fr) * 32 + 16 + 8 * fh);
#pragma unroll
    for (int j = 0; j < 4; j++) {
      bfr[j].q[0] = *(const uint4*)(cW + (j * 16 + fr) * 32 + 8 * fh);
      bfr[j].q[1] = *(const uint4*)(cW + (j * 16 + fr) * 32 + 16 + 8 * fh);
    }
#pragma unroll
    for (int j = 0; j < 4; j++)
      acc[j] = __builtin_amdgcn_wmma_f32_16x16x32_bf16(
          false, a.v, false, bfr[j].v, (short)0, acc[j], false, false);
    __syncthreads();                          // all reads done before overwrite
  }
}

#define GEMM_LDS \
  __shared__ __align__(16) unsigned short ldsA[2 * 128 * 32]; \
  __shared__ __align__(16) unsigned short ldsW[2 * 64 * 32];

// ---------------- GEMM 1: QKV + bias, scatter to per-head q/k/v (q scaled) --
__global__ void __launch_bounds__(256)
gemm_qkv(const unsigned short* __restrict__ A, const unsigned short* __restrict__ WT,
         const float* __restrict__ bias, unsigned short* __restrict__ qb,
         unsigned short* __restrict__ kb, unsigned short* __restrict__ vb) {
  GEMM_LDS
  v8f acc[4];
  int rowBase = blockIdx.x * 128, nBase = blockIdx.y * 64;
  gemm_main(A, WT, C_, rowBase, nBase, ldsA, ldsW, acc);
  const int lane = threadIdx.x & 31, wid = threadIdx.x >> 5;
  const int fr = lane & 15, fh = lane >> 4;
#pragma unroll
  for (int j = 0; j < 4; j++) {
    int n = nBase + j * 16 + fr;
    int part = n / 192, head = (n % 192) / 32, d = n & 31;
    unsigned short* dst = (part == 0) ? qb : ((part == 1) ? kb : vb);
    float sc = (part == 0) ? SCALE_ : 1.f;
    float bv = bias[n];
#pragma unroll
    for (int i = 0; i < 8; i++) {
      int r = rowBase + wid * 16 + i + 8 * fh;
      int win = r / TW_, t = r - win * TW_;
      float val = (acc[j][i] + bv) * sc;
      dst[(((size_t)win * NH_ + head) * TW_ + t) * HD_ + d] = f32_bf16(val);
    }
  }
}

// ---------------- attention per (window, head): S=qk^T+bias+mask; O=S@v -----
__global__ void __launch_bounds__(128)
attn_kernel(const unsigned short* __restrict__ qb, const unsigned short* __restrict__ kb,
            const unsigned short* __restrict__ vb, const float* __restrict__ pos_bias,
            unsigned short* __restrict__ ob) {
  __shared__ __align__(16) unsigned short ldsV[32 * 64];   // v^T, zero padded
  __shared__ __align__(16) unsigned short ldsS[64 * 64];   // S in bf16
  int wh_id = blockIdx.x;                 // win*NH + head
  int win = wh_id / NH_, head = wh_id - win * NH_;
  int tid = threadIdx.x, lane = tid & 31, wid = tid >> 5;
  const size_t base = (size_t)wh_id * TW_ * HD_;

  for (int idx = tid; idx < 32 * 64; idx += 128) ldsV[idx] = 0;
  __syncthreads();
  for (int idx = tid; idx < TW_ * HD_; idx += 128) {
    int t = idx >> 5, d = idx & 31;
    ldsV[d * 64 + t] = vb[base + idx];
  }
  __syncthreads();

  const int fr = lane & 15, fh = lane >> 4;
  Frag a;
  {
    int t = wid * 16 + fr;
    if (t < TW_) {
      const unsigned short* qr = qb + base + (size_t)t * HD_;
      a.q[0] = *(const uint4*)(qr + 8 * fh);
      a.q[1] = *(const uint4*)(qr + 16 + 8 * fh);
    } else { a.q[0] = make_uint4(0,0,0,0); a.q[1] = make_uint4(0,0,0,0); }
  }
  int w_img = win & 63;
  int mwh = w_img >> 3, mww = w_img & 7;

#pragma unroll
  for (int jt = 0; jt < 4; jt++) {
    Frag b;
    int nrow = jt * 16 + fr;
    if (nrow < TW_) {
      const unsigned short* kr = kb + base + (size_t)nrow * HD_;
      b.q[0] = *(const uint4*)(kr + 8 * fh);
      b.q[1] = *(const uint4*)(kr + 16 + 8 * fh);
    } else { b.q[0] = make_uint4(0,0,0,0); b.q[1] = make_uint4(0,0,0,0); }
    v8f acc; for (int e = 0; e < 8; e++) acc[e] = 0.f;
    acc = __builtin_amdgcn_wmma_f32_16x16x32_bf16(
        false, a.v, false, b.v, (short)0, acc, false, false);
    int gj = jt * 16 + fr;
#pragma unroll
    for (int i = 0; i < 8; i++) {
      int gi = wid * 16 + i + 8 * fh;
      unsigned short sv = 0;
      if (gi < TW_ && gj < TW_) {
        int ih = gi / 7, iw = gi - ih * 7;
        int jh = gj / 7, jw = gj - jh * 7;
        float bias = pos_bias[((ih - jh + 6) * 13 + (iw - jw + 6)) * NH_ + head];
        int ghi = mwh * 7 + ih, gwi = mww * 7 + iw;
        int ghj = mwh * 7 + jh, gwj = mww * 7 + jw;
        int ci = (ghi < 49 ? 0 : (ghi < 53 ? 1 : 2)) * 3 + (gwi < 49 ? 0 : (gwi < 53 ? 1 : 2));
        int cj = (ghj < 49 ? 0 : (ghj < 53 ? 1 : 2)) * 3 + (gwj < 49 ? 0 : (gwj < 53 ? 1 : 2));
        float m = (ci == cj) ? 0.f : -100.f;
        sv = f32_bf16(acc[i] + bias + m);
      }
      ldsS[(wid * 16 + i + 8 * fh) * 64 + gj] = sv;
    }
  }
  // O = S @ v  (each wave consumes only its own S rows; LDS in-order per wave)
#pragma unroll
  for (int ot = 0; ot < 2; ot++) {
    v8f acc; for (int e = 0; e < 8; e++) acc[e] = 0.f;
#pragma unroll
    for (int ks = 0; ks < 2; ks++) {
      Frag sa, vf;
      const unsigned short* srow = ldsS + (wid * 16 + fr) * 64 + 32 * ks;
      sa.q[0] = *(const uint4*)(srow + 8 * fh);
      sa.q[1] = *(const uint4*)(srow + 16 + 8 * fh);
      const unsigned short* vrow = ldsV + (ot * 16 + fr) * 64 + 32 * ks;
      vf.q[0] = *(const uint4*)(vrow + 8 * fh);
      vf.q[1] = *(const uint4*)(vrow + 16 + 8 * fh);
      acc = __builtin_amdgcn_wmma_f32_16x16x32_bf16(
          false, sa.v, false, vf.v, (short)0, acc, false, false);
    }
#pragma unroll
    for (int i = 0; i < 8; i++) {
      int t = wid * 16 + i + 8 * fh;
      if (t < TW_) {
        int d = ot * 16 + fr;
        ob[((size_t)win * TW_ + t) * C_ + head * HD_ + d] = f32_bf16(acc[i]);
      }
    }
  }
}

// ---------------- GEMM 2: proj + bias, window-reverse + unshift + residual --
__global__ void __launch_bounds__(256)
gemm_proj(const unsigned short* __restrict__ A, const unsigned short* __restrict__ WT,
          const float* __restrict__ bias, const float* __restrict__ xin,
          float* __restrict__ x1) {
  GEMM_LDS
  v8f acc[4];
  int rowBase = blockIdx.x * 128, nBase = blockIdx.y * 64;
  gemm_main(A, WT, C_, rowBase, nBase, ldsA, ldsW, acc);
  const int lane = threadIdx.x & 31, wid = threadIdx.x >> 5;
  const int fr = lane & 15, fh = lane >> 4;
#pragma unroll
  for (int j = 0; j < 4; j++) {
    int n = nBase + j * 16 + fr;
    float bv = bias[n];
#pragma unroll
    for (int i = 0; i < 8; i++) {
      int r = rowBase + wid * 16 + i + 8 * fh;
      int win = r / TW_, t = r - win * TW_;
      int bimg = win >> 6, w = win & 63;
      int wh = w >> 3, ww = w & 7;
      int ih = t / 7, iw = t - ih * 7;
      int oh = (wh * 7 + ih + SS_) % HH;      // reverse shift (+SS)
      int ow = (ww * 7 + iw + SS_) % WWID;
      size_t o = ((size_t)bimg * T_ + oh * WWID + ow) * C_ + n;
      x1[o] = xin[o] + acc[j][i] + bv;
    }
  }
}

// ---------------- GEMM 3: MLP fc1 + bias + exact GELU -> bf16 ---------------
__global__ void __launch_bounds__(256)
gemm_mlp1(const unsigned short* __restrict__ A, const unsigned short* __restrict__ WT,
          const float* __restrict__ bias, unsigned short* __restrict__ m1) {
  GEMM_LDS
  v8f acc[4];
  int rowBase = blockIdx.x * 128, nBase = blockIdx.y * 64;
  gemm_main(A, WT, C_, rowBase, nBase, ldsA, ldsW, acc);
  const int lane = threadIdx.x & 31, wid = threadIdx.x >> 5;
  const int fr = lane & 15, fh = lane >> 4;
#pragma unroll
  for (int j = 0; j < 4; j++) {
    int n = nBase + j * 16 + fr;
    float bv = bias[n];
#pragma unroll
    for (int i = 0; i < 8; i++) {
      int r = rowBase + wid * 16 + i + 8 * fh;
      float v = acc[j][i] + bv;
      v = 0.5f * v * (1.f + erff(v * 0.70710678118654752f));
      m1[(size_t)r * 768 + n] = f32_bf16(v);
    }
  }
}

// ---------------- GEMM 4: MLP fc2 + bias + residual -> f32 output -----------
__global__ void __launch_bounds__(256)
gemm_mlp2(const unsigned short* __restrict__ A, const unsigned short* __restrict__ WT,
          const float* __restrict__ bias, const float* __restrict__ x1,
          float* __restrict__ out) {
  GEMM_LDS
  v8f acc[4];
  int rowBase = blockIdx.x * 128, nBase = blockIdx.y * 64;
  gemm_main(A, WT, 768, rowBase, nBase, ldsA, ldsW, acc);
  const int lane = threadIdx.x & 31, wid = threadIdx.x >> 5;
  const int fr = lane & 15, fh = lane >> 4;
#pragma unroll
  for (int j = 0; j < 4; j++) {
    int n = nBase + j * 16 + fr;
    float bv = bias[n];
#pragma unroll
    for (int i = 0; i < 8; i++) {
      int r = rowBase + wid * 16 + i + 8 * fh;
      size_t o = (size_t)r * C_ + n;
      out[o] = x1[o] + acc[j][i] + bv;
    }
  }
}

// ---------------- launcher ----------------
extern "C" void kernel_launch(void* const* d_in, const int* in_sizes, int n_in,
                              void* d_out, int out_size, void* d_ws, size_t ws_size,
                              hipStream_t stream) {
  (void)in_sizes; (void)n_in; (void)out_size; (void)ws_size;
  const float* x      = (const float*)d_in[0];
  const float* ln1_g  = (const float*)d_in[1];
  const float* ln1_b  = (const float*)d_in[2];
  const float* qkv_w  = (const float*)d_in[3];
  const float* qkv_b  = (const float*)d_in[4];
  const float* pos_b  = (const float*)d_in[5];
  const float* proj_w = (const float*)d_in[6];
  const float* proj_b = (const float*)d_in[7];
  const float* ln2_g  = (const float*)d_in[8];
  const float* ln2_b  = (const float*)d_in[9];
  const float* w1     = (const float*)d_in[10];
  const float* b1     = (const float*)d_in[11];
  const float* w2     = (const float*)d_in[12];
  const float* b2     = (const float*)d_in[13];
  float* out = (float*)d_out;

  char* ws = (char*)d_ws;
  size_t off = 0;
  auto alloc = [&](size_t bytes) -> char* {
    char* p = ws + off;
    off = (off + bytes + 255) & ~(size_t)255;
    return p;
  };
  unsigned short* wt_qkv = (unsigned short*)alloc((size_t)576 * 192 * 2);
  unsigned short* wt_prj = (unsigned short*)alloc((size_t)192 * 192 * 2);
  unsigned short* wt_m1  = (unsigned short*)alloc((size_t)768 * 192 * 2);
  unsigned short* wt_m2  = (unsigned short*)alloc((size_t)192 * 768 * 2);
  unsigned short* hB = (unsigned short*)alloc((size_t)MTOT * C_ * 2);   // also h2
  unsigned short* qB = (unsigned short*)alloc((size_t)MTOT * C_ * 2);   // m1 overlays q..O
  unsigned short* kB = (unsigned short*)alloc((size_t)MTOT * C_ * 2);
  unsigned short* vB = (unsigned short*)alloc((size_t)MTOT * C_ * 2);
  unsigned short* oB = (unsigned short*)alloc((size_t)MTOT * C_ * 2);
  float*          x1 = (float*)alloc((size_t)MTOT * C_ * 4);
  unsigned short* m1 = qB;    // q/k/v/o dead after proj; 4*38.5MB == MTOT*768*2
  unsigned short* h2 = hB;    // hB dead after QKV gemm

  // 0) weight convert/transpose
  wconv_kernel<<<(192 * 576 + 255) / 256, 256, 0, stream>>>(qkv_w, wt_qkv, 192, 576);
  wconv_kernel<<<(192 * 192 + 255) / 256, 256, 0, stream>>>(proj_w, wt_prj, 192, 192);
  wconv_kernel<<<(192 * 768 + 255) / 256, 256, 0, stream>>>(w1, wt_m1, 192, 768);
  wconv_kernel<<<(768 * 192 + 255) / 256, 256, 0, stream>>>(w2, wt_m2, 768, 192);

  // 1) LN1 + cyclic shift + window partition -> bf16
  ln_kernel<<<MTOT / 8, 256, 0, stream>>>(x, ln1_g, ln1_b, hB, 1);

  // 2) QKV GEMM (M=100352, N=576, K=192)
  gemm_qkv<<<dim3(MTOT / 128, 576 / 64), 256, 0, stream>>>(hB, wt_qkv, qkv_b, qB, kB, vB);

  // 3) window attention (no softmax, per reference)
  attn_kernel<<<NWIN * NH_, 128, 0, stream>>>(qB, kB, vB, pos_b, oB);

  // 4) proj GEMM + reverse windows + reverse shift + residual -> x1 (f32)
  gemm_proj<<<dim3(MTOT / 128, 192 / 64), 256, 0, stream>>>(oB, wt_prj, proj_b, x, x1);

  // 5) LN2 -> bf16
  ln_kernel<<<MTOT / 8, 256, 0, stream>>>(x1, ln2_g, ln2_b, h2, 0);

  // 6) MLP fc1 + GELU (N=768, K=192)
  gemm_mlp1<<<dim3(MTOT / 128, 768 / 64), 256, 0, stream>>>(h2, wt_m1, b1, m1);

  // 7) MLP fc2 + residual -> out (N=192, K=768)
  gemm_mlp2<<<dim3(MTOT / 128, 192 / 64), 256, 0, stream>>>(m1, wt_m2, b2, x1, out);
}
